// TTpoly4dim_17583596110585
// MI455X (gfx1250) — matrix operand order repaired and checked by reading, not compiled, not run on hardware
//
#include <hip/hip_runtime.h>
#include <hip/hip_bf16.h>

// TT-polynomial regression, 4 dims, deg=7 (d1=8), rank=8, B=2e6.
// Reformulated as Khatri-Rao feature GEMMs -> V_WMMA_F32_16X16X32_BF16.
// Per 16-element batch tile (one wave): 7 WMMAs, LDS-staged transposes.

typedef __attribute__((ext_vector_type(16))) __bf16 v16bf;
typedef __attribute__((ext_vector_type(8)))  float  v8f;
typedef __attribute__((ext_vector_type(4)))  float  v4f;

#define COLS 20  // padded LDS column stride (floats): bank-conflict-free, 16B aligned

__device__ __forceinline__ __bf16 bf(float f) { return (__bf16)f; }

// Store D (16x16 f32, column-striped across lanes) transposed into LDS so it
// becomes row-addressable for the next stage's A-fragment build.
// Lane holds D[m = v + 8h][n] in c[v]  ->  tbuf[n*COLS + 8h + v] (8 consecutive
// floats per lane => two ds_store_b128).
__device__ __forceinline__ void transpose_store(float* tcol, v8f c) {
  v4f lo, hi;
  lo.x = c[0]; lo.y = c[1]; lo.z = c[2]; lo.w = c[3];
  hi.x = c[4]; hi.y = c[5]; hi.z = c[6]; hi.w = c[7];
  *(v4f*)tcol       = lo;
  *(v4f*)(tcol + 4) = hi;
}

// One TT core: t'[m,s] = sum_{r,d} t[m,r] * Pk[m,d] * G[r,d,s]
//            = (Z @ Gflat),  Z[m, r*8+d] = t[m,r]*Pk[m,d],  K = 64 -> 2 WMMAs.
// Lane (row n, half h) covers K slices {8h+j, 16+8h+j} (+32 for 2nd fragment),
// i.e. r in {h, 2+h} and {4+h, 6+h}; t read back from the LDS transpose.
__device__ __forceinline__ v8f tt_stage(const float* trow, const float* Pk,
                                        int n, int h, v16bf bA, v16bf bB) {
  const float tA = trow[(h + 0) * COLS + n];
  const float tB = trow[(h + 2) * COLS + n];
  const float tC = trow[(h + 4) * COLS + n];
  const float tD = trow[(h + 6) * COLS + n];
  v16bf aA, aB;
#pragma unroll
  for (int j = 0; j < 8; ++j) {
    aA[j]     = bf(tA * Pk[j]);
    aA[j + 8] = bf(tB * Pk[j]);
    aB[j]     = bf(tC * Pk[j]);
    aB[j + 8] = bf(tD * Pk[j]);
  }
  v8f c = {};
  c = __builtin_amdgcn_wmma_f32_16x16x32_bf16(false, aA, false, bA, (short)0, c, false, false);
  c = __builtin_amdgcn_wmma_f32_16x16x32_bf16(false, aB, false, bB, (short)0, c, false, false);
  return c;
}

__global__ __launch_bounds__(256) void tt_poly_wmma(
    const float* __restrict__ X,
    const float* __restrict__ G0, const float* __restrict__ G1,
    const float* __restrict__ G2, const float* __restrict__ G3,
    float* __restrict__ Out, int nElem)
{
  __shared__ __align__(16) float tbuf[8][16 * COLS];  // per-wave transpose scratch
  __shared__ float rbuf[8][16];                       // per-wave result gather

  const int lane = threadIdx.x & 31;
  const int wib  = threadIdx.x >> 5;   // wave in block (8 waves / 256 threads)
  const int n    = lane & 15;          // fragment column == this lane's batch row
  const int h    = lane >> 4;          // wave half (K-slice selector)
  const int waveId = blockIdx.x * (blockDim.x >> 5) + wib;
  const int nWaves = gridDim.x * (blockDim.x >> 5);
  const int nTiles = (nElem + 15) >> 4;

  // ---- Loop-invariant weight (B operand) fragments, bf16, kept in VGPRs ----
  // B 32x16 layout: lane n holds column n, K = 8h + j (j<8), 16 + 8h + (j-8) (j>=8).
  v16bf b0{}, b1a{}, b1b{}, b2a{}, b2b{}, b3a{}, b3b{};
#pragma unroll
  for (int j = 0; j < 16; ++j) {
    const int k = 8 * h + j + (j >= 8 ? 8 : 0);
    if (n < 8) {
      if (h == 0 && j < 8) b0[j] = bf(G0[j * 8 + n]);   // G0[d][r], K-padded w/ 0
      b1a[j] = bf(G1[k * 8 + n]);                       // G1flat[k][s], k=0..31
      b1b[j] = bf(G1[(k + 32) * 8 + n]);                // k=32..63
      b2a[j] = bf(G2[k * 8 + n]);
      b2b[j] = bf(G2[(k + 32) * 8 + n]);
    }
    if (n == 0) {                                       // final core: N=1 column
      b3a[j] = bf(G3[k]);
      b3b[j] = bf(G3[k + 32]);
    }
  }

  float*       tcol = &tbuf[wib][n * COLS + 8 * h];
  const float* trow = &tbuf[wib][0];

  for (int tile = waveId; tile < nTiles; tile += nWaves) {
    const int elem  = tile * 16 + n;                 // halves duplicate the row
    const int eclmp = elem < nElem ? elem : nElem - 1;

    // Stream-once read: NT hint, don't rinse L2. Prefetch next tile's X.
    const v4f x = __builtin_nontemporal_load((const v4f*)X + eclmp);
    if (tile + nWaves < nTiles)
      __builtin_prefetch((const v4f*)X + (size_t)(tile + nWaves) * 16 + n, 0, 0);

    // Monomial bases x_d^0..7 (f32; bf16-converted at fragment build).
    float P0[8], P1[8], P2[8], P3[8];
    P0[0] = P1[0] = P2[0] = P3[0] = 1.0f;
#pragma unroll
    for (int j = 1; j < 8; ++j) {
      P0[j] = P0[j - 1] * x.x;
      P1[j] = P1[j - 1] * x.y;
      P2[j] = P2[j - 1] * x.z;
      P3[j] = P3[j - 1] * x.w;
    }

    // ---- stage 0: t0 = P0 . G0  (K=0..7 live, rest zero-padded) ----
    v16bf a0{};
    if (h == 0) {
#pragma unroll
      for (int j = 0; j < 8; ++j) a0[j] = bf(P0[j]);
    }
    v8f c0 = {};
    c0 = __builtin_amdgcn_wmma_f32_16x16x32_bf16(false, a0, false, b0, (short)0, c0, false, false);
    transpose_store(tcol, c0);

    // ---- stage 1, stage 2: middle TT cores (2 WMMAs each) ----
    v8f c1 = tt_stage(trow, P1, n, h, b1a, b1b);
    transpose_store(tcol, c1);
    v8f c2 = tt_stage(trow, P2, n, h, b2a, b2b);
    transpose_store(tcol, c2);

    // ---- stage 3: res[m] = Z3[m,:] . G3flat  (N=1 column of D) ----
    v8f c3 = tt_stage(trow, P3, n, h, b3a, b3b);

    // D[m][0] lives in lanes 0 (m=0..7) and 16 (m=8..15); gather via LDS.
    if (n == 0) {
#pragma unroll
      for (int v = 0; v < 8; ++v) rbuf[wib][8 * h + v] = c3[v];
    }
    const float res = rbuf[wib][n];
    if (h == 0 && elem < nElem)
      __builtin_nontemporal_store(res, Out + elem);
  }
}

extern "C" void kernel_launch(void* const* d_in, const int* in_sizes, int n_in,
                              void* d_out, int out_size, void* d_ws, size_t ws_size,
                              hipStream_t stream) {
  const float* X  = (const float*)d_in[0];
  const float* G0 = (const float*)d_in[1];
  const float* G1 = (const float*)d_in[2];
  const float* G2 = (const float*)d_in[3];
  const float* G3 = (const float*)d_in[4];
  float* Out = (float*)d_out;

  const int nElem  = in_sizes[0] / 4;        // X is [B,4] f32
  const int nTiles = (nElem + 15) >> 4;

  int blocks = 1024;                         // 8192 waves, grid-stride over tiles
  const int maxUseful = (nTiles + 7) / 8;    // >=1 tile per wave
  if (blocks > maxUseful) blocks = maxUseful > 0 ? maxUseful : 1;

  tt_poly_wmma<<<blocks, 256, 0, stream>>>(X, G0, G1, G2, G3, Out, nElem);
}
